// TTLinear_19619410608255
// MI455X (gfx1250) — matrix operand order repaired and checked by reading, not compiled, not run on hardware
//
#include <hip/hip_runtime.h>
#include <hip/hip_bf16.h>

// TT-linear 4096->4096, B=2048, ranks (1,16,16,16,1), modes 8^4.
// Fused 2-stage formulation:
//   B23[(i2,i3),(r2,m2,m3)] = sum_r3 core2[(r2,m2),(i2,r3)] * core3[(r3,m3),i3]
//   T01[(m0,m1),(i0,i1,r2)] = sum_r1 core0[m0,(i0,r1)]      * core1[(r1,m1),(i1,r2)]
// Per batch b:
//   zA[(i0,i1),(r2,m2,m3)] = X_b[(i0,i1),(i2,i3)] @ B23          (64x64 * 64x1024)
//   reinterpret zA as zB[(i0,i1,r2),(m2,m3)]   (free reshape)
//   out_b[(m0,m1),(m2,m3)] = T01 @ zB + bias                      (64x1024 * 1024x64)
// z kept entirely in LDS (bf16, ~129 KB) -> no HBM intermediates.

typedef __attribute__((ext_vector_type(16))) __bf16 v16bf;
typedef __attribute__((ext_vector_type(8)))  float  v8f;

#define Z_STRIDE 1032   // 1024 + 8 pad (bf16 elems) -> conflict-free ds_b128
#define X_STRIDE 72     // 64 + 8 pad

__device__ __forceinline__ unsigned short f2bf(float f) {
  unsigned int u = __builtin_bit_cast(unsigned int, f);
  u += 0x7FFFu + ((u >> 16) & 1u);          // round-to-nearest-even
  return (unsigned short)(u >> 16);
}

struct Frag32B { uint4 lo, hi; };

__device__ __forceinline__ v16bf ld_frag(const unsigned short* p) {
  Frag32B t;
  t.lo = *(const uint4*)(p);
  t.hi = *(const uint4*)(p + 8);
  return __builtin_bit_cast(v16bf, t);
}

// ---- prep: B23 in per-lane WMMA B-fragment order ----------------------------
// layout: B23f[((nt*2 + kt)*32 + lane)*16 + e]
//   value at K = 32*kt + (lane>>4)*16 + e  (K = i2*8+i3),  N = 16*nt + (lane&15)
__global__ void tt_prep_b23(const float* __restrict__ core2,
                            const float* __restrict__ core3,
                            unsigned short* __restrict__ B23f) {
  int t    = blockIdx.x * blockDim.x + threadIdx.x;   // 65536 threads
  int e    = t & 15;
  int lane = (t >> 4) & 31;
  int kt   = (t >> 9) & 1;
  int nt   = t >> 10;
  int k = 32 * kt + ((lane >> 4) << 4) + e;           // (i2,i3)
  int n = 16 * nt + (lane & 15);                      // (r2,m2,m3)
  int i2 = k >> 3, i3 = k & 7;
  int r2 = n >> 6, m2 = (n >> 3) & 7, m3 = n & 7;
  float acc = 0.f;
#pragma unroll
  for (int r3 = 0; r3 < 16; ++r3)
    acc += core2[(r2 * 8 + m2) * 128 + i2 * 16 + r3] *
           core3[(r3 * 8 + m3) * 8 + i3];
  B23f[t] = f2bf(acc);
}

// ---- prep: T01 in per-lane WMMA A-fragment order ----------------------------
// layout: T01f[((mt*32 + kt)*32 + lane)*16 + e]
//   M = 16*mt + (lane&15),  K = 32*kt + (e&8?16:0) + (lane>>4)*8 + (e&7)
__global__ void tt_prep_t01(const float* __restrict__ core0,
                            const float* __restrict__ core1,
                            unsigned short* __restrict__ T01f) {
  int t    = blockIdx.x * blockDim.x + threadIdx.x;   // 65536 threads
  int e    = t & 15;
  int lane = (t >> 4) & 31;
  int kt   = (t >> 9) & 31;
  int mt   = t >> 14;
  int m = 16 * mt + (lane & 15);                      // (m0,m1)
  int k = 32 * kt + ((e & 8) ? 16 : 0) + ((lane >> 4) << 3) + (e & 7); // (i0,i1,r2)
  int m0 = m >> 3, m1 = m & 7;
  int i0 = k >> 7, i1 = (k >> 4) & 7, r2 = k & 15;
  float acc = 0.f;
#pragma unroll
  for (int r1 = 0; r1 < 16; ++r1)
    acc += core0[m0 * 128 + i0 * 16 + r1] *
           core1[(r1 * 8 + m1) * 128 + i1 * 16 + r2];
  T01f[t] = f2bf(acc);
}

// ---- fused main kernel: one workgroup (8 wave32) per batch element ----------
__global__ __launch_bounds__(256)
void tt_fused_kernel(const float* __restrict__ x,
                     const unsigned short* __restrict__ B23f,
                     const unsigned short* __restrict__ T01f,
                     const float* __restrict__ bias,
                     float* __restrict__ out) {
  extern __shared__ unsigned short smem[];
  unsigned short* zlds = smem;                    // [64 n=(m2,m3)][Z_STRIDE k=(i0,i1,r2)]
  unsigned short* xlds = smem + 64 * Z_STRIDE;    // [64 (i0,i1)][X_STRIDE (i2,i3)]

  const int b     = blockIdx.x;
  const int tid   = threadIdx.x;
  const int lane  = tid & 31;
  // wave id as a compiler-visible scalar: avoids exec-masked "divergent" loops
  // (WMMA requires EXEC all-1s; scalar loop bounds keep control flow uniform)
  const int w     = __builtin_amdgcn_readfirstlane(tid >> 5);
  const int ln15  = lane & 15;
  const int khalf = lane >> 4;

  // ---- stage x: global f32 -> LDS bf16, [64 x 64] row-major, padded --------
  {
    const float4* xg = (const float4*)(x + (size_t)b * 4096);
#pragma unroll
    for (int i = 0; i < 4; ++i) {
      int idx = tid + 256 * i;                    // float4 index 0..1023
      int r   = idx >> 4;                         // row (i0,i1)
      int c4  = idx & 15;                         // float4 within row
      float4 v = xg[idx];
      uint2 p;
      p.x = (unsigned)f2bf(v.x) | ((unsigned)f2bf(v.y) << 16);
      p.y = (unsigned)f2bf(v.z) | ((unsigned)f2bf(v.w) << 16);
      *(uint2*)(xlds + r * X_STRIDE + c4 * 4) = p;
    }
  }
  __syncthreads();

  // ---- stage A: z = X(64x64) @ B23(64x1024), scatter bf16 into zlds --------
  {
    const int mt     = w & 3;
    const int ntBase = (w >> 2) * 32;             // 2 waves x 32 nt cover N=1024
    const unsigned short* arow = xlds + (16 * mt + ln15) * X_STRIDE;
    v16bf a0, a1;
    {
      Frag32B t;
      t.lo = *(const uint4*)(arow + khalf * 16);          // K = 16*khalf + 0..7
      t.hi = *(const uint4*)(arow + 8 + khalf * 16);      // K = 16*khalf + 8..15
      a0 = __builtin_bit_cast(v16bf, t);
      t.lo = *(const uint4*)(arow + 32 + khalf * 16);
      t.hi = *(const uint4*)(arow + 40 + khalf * 16);
      a1 = __builtin_bit_cast(v16bf, t);
    }
#pragma unroll 2
    for (int nto = 0; nto < 32; ++nto) {
      const int nt = ntBase + nto;
      const unsigned short* bp = B23f + (size_t)((nt * 2) * 32 + lane) * 16;
      v16bf b0 = ld_frag(bp);
      v16bf b1 = ld_frag(bp + 512);               // kt = 1
      v8f c = {};
      c = __builtin_amdgcn_wmma_f32_16x16x32_bf16(false, a0, false, b0, (short)0, c, false, false);
      c = __builtin_amdgcn_wmma_f32_16x16x32_bf16(false, a1, false, b1, (short)0, c, false, false);
      // D elem v: row M = v + 8*khalf (i0,i1 in tile), col N = ln15
      // z_B row kB = (16*mt + M)*16 + r2 ; z_B col mm = (m2,m3)
      const int r2t = nt >> 2;                    // constant within tile
      const int mm  = (nt & 3) * 16 + ln15;
      unsigned short* zp = zlds + mm * Z_STRIDE + 256 * mt + 128 * khalf + r2t;
      // NOTE: keep the explicit RNE f2bf form here. The "(bits+0x8000)>>16"
      // shortcut triggers a backend store-merging bug (all 8 d16_hi stores
      // sourced the element-0 register).
#pragma unroll
      for (int v = 0; v < 8; ++v)
        zp[16 * v] = f2bf(c[v]);
    }
  }
  __syncthreads();

  // ---- stage B: out = T01(64x1024) @ zB(1024x64), K-loop of 32 WMMAs -------
  {
    const int mt  = w & 3;
    const int ntB = (w >> 2) * 2;                 // each wave: 2 N-tiles
    const unsigned short* zb0 = zlds + (16 * ntB + ln15) * Z_STRIDE + 8 * khalf;
    const unsigned short* zb1 = zb0 + 16 * Z_STRIDE;
    const unsigned short* ap  = T01f + (size_t)((mt * 32) * 32 + lane) * 16;
    v8f acc0 = {}, acc1 = {};
#pragma unroll
    for (int kt = 0; kt < 32; ++kt) {
      v16bf a = ld_frag(ap + kt * 512);
      Frag32B t;
      t.lo = *(const uint4*)(zb0 + 32 * kt);
      t.hi = *(const uint4*)(zb0 + 32 * kt + 16);
      v16bf b0 = __builtin_bit_cast(v16bf, t);
      t.lo = *(const uint4*)(zb1 + 32 * kt);
      t.hi = *(const uint4*)(zb1 + 32 * kt + 16);
      v16bf b1 = __builtin_bit_cast(v16bf, t);
      acc0 = __builtin_amdgcn_wmma_f32_16x16x32_bf16(false, a, false, b0, (short)0, acc0, false, false);
      acc1 = __builtin_amdgcn_wmma_f32_16x16x32_bf16(false, a, false, b1, (short)0, acc1, false, false);
    }
    // epilogue: out[b, (m0,m1)*64 + (m2,m3)] = acc + bias
    float* op = out + (size_t)b * 4096;
    const int n0 = 16 * ntB + ln15;
    const int n1 = n0 + 16;
#pragma unroll
    for (int v = 0; v < 8; ++v) {
      int row = 16 * mt + v + 8 * khalf;          // (m0,m1)
      op[row * 64 + n0] = acc0[v] + bias[row * 64 + n0];
      op[row * 64 + n1] = acc1[v] + bias[row * 64 + n1];
    }
  }
}

extern "C" void kernel_launch(void* const* d_in, const int* in_sizes, int n_in,
                              void* d_out, int out_size, void* d_ws, size_t ws_size,
                              hipStream_t stream) {
  const float* x     = (const float*)d_in[0];
  const float* core0 = (const float*)d_in[1];
  const float* core1 = (const float*)d_in[2];
  const float* core2 = (const float*)d_in[3];
  const float* core3 = (const float*)d_in[4];
  const float* bias  = (const float*)d_in[5];
  float* out = (float*)d_out;

  unsigned short* B23f = (unsigned short*)d_ws;   // 64*1024 bf16 = 128 KB
  unsigned short* T01f = B23f + 65536;            // 64*1024 bf16 = 128 KB

  tt_prep_b23<<<256, 256, 0, stream>>>(core2, core3, B23f);
  tt_prep_t01<<<256, 256, 0, stream>>>(core0, core1, T01f);

  const size_t lds_bytes = (size_t)(64 * Z_STRIDE + 64 * X_STRIDE) * sizeof(unsigned short); // ~141 KB
  tt_fused_kernel<<<2048, 256, lds_bytes, stream>>>(x, B23f, T01f, bias, out);
}
